// GraphWaveletNeuralNetwork_55422257988357
// MI455X (gfx1250) — matrix operand order repaired and checked by reading, not compiled, not run on hardware
//
#include <hip/hip_runtime.h>

// ---- vector types for WMMA (wave32, CDNA5) ----
typedef __attribute__((ext_vector_type(16))) _Float16 v16h;
typedef __attribute__((ext_vector_type(8)))  _Float16 v8h;
typedef __attribute__((ext_vector_type(4)))  _Float16 v4h;
typedef __attribute__((ext_vector_type(8)))  float    v8f;
typedef __attribute__((ext_vector_type(4)))  float    v4f;   // native vec4 (builtin-compatible)

#define BM 64              // rows of C per block
#define BN 128             // cols of C per block (== C_OUT, full width)
#define BK 32              // K chunk (== WMMA K for f16)
#define LDSS 40            // f16 elems per LDS row: 32 + 8 pad -> 80B = 20 dword banks

// C[M x 128] = A[M x K] @ B[K x 128], optionally scaled per output row.
// A row-major (lda = K), B row-major (ldb = 128), C row-major (ldc = 128).
__global__ __launch_bounds__(256)
void gwnn_gemm_f16(const float* __restrict__ A, int lda, int K,
                   const float* __restrict__ B,
                   float* __restrict__ C,
                   const float* __restrict__ row_scale)
{
    __shared__ __attribute__((aligned(16))) _Float16 As[BM * LDSS]; // [row][k]
    __shared__ __attribute__((aligned(16))) _Float16 Bs[BN * LDSS]; // [col][k] (B^T)

    const int tid  = threadIdx.x;
    const int lane = tid & 31;
    const int wave = tid >> 5;       // 0..7
    const int wrow = wave >> 1;      // 0..3 : 16-row group
    const int wcol = wave & 1;       // 0..1 : 64-col group
    const int half = lane >> 4;      // 0/1  : lane half (ISA fragment layout)
    const int l16  = lane & 15;

    const int row0 = blockIdx.x * BM;

    // B staging: each thread owns one 4x4 (k x col) micro-tile of the chunk
    const int bcb = (tid & 31) * 4;  // col base   (32 thr * 4 = 128 cols)
    const int bkb = (tid >> 5) * 4;  // k base     ( 8 thr * 4 = 32 k)

    // Per-thread 32-bit BYTE offsets, computed once. Chunk advance goes into
    // the (uniform) scalar base pointer -> saddr-form global loads, no
    // per-iteration 64-bit VALU address math. Max offset: 8191*8192*4 < 2^31.
    uint32_t aoff[2], boff[4];
#pragma unroll
    for (int i = 0; i < 2; ++i) {
        const int idx = tid + i * 256;
        const int r = idx >> 3, k4 = idx & 7;
        aoff[i] = (uint32_t)(((row0 + r) * lda + k4 * 4) * 4);
    }
#pragma unroll
    for (int j = 0; j < 4; ++j)
        boff[j] = (uint32_t)(((bkb + j) * BN + bcb) * 4);

    v4f ra[2];   // A prefetch regs (64 rows * 8 float4 = 512 -> 2/thread)
    v4f rb[4];   // B prefetch regs (one 4x4 micro-tile)

    const int NK = K / BK;
    const char* aBase = (const char*)A;   // advances BK*4 bytes per chunk
    const char* bBase = (const char*)B;   // advances BK*BN*4 bytes per chunk

    auto loadA = [&](const char* ab) {
        // A is the once-streamed big matrix: non-temporal, keep L2 for B panel
#pragma unroll
        for (int i = 0; i < 2; ++i)
            ra[i] = __builtin_nontemporal_load((const v4f*)(ab + aoff[i]));
    };
    auto loadB = [&](const char* bb) {
#pragma unroll
        for (int j = 0; j < 4; ++j)
            rb[j] = *(const v4f*)(bb + boff[j]);
    };
    auto stage = [&]() {
        // A: f32 -> f16, ds_store_b64, row-major [row][k]
#pragma unroll
        for (int i = 0; i < 2; ++i) {
            const int idx = tid + i * 256;
            const int r = idx >> 3, k4 = idx & 7;
            v4h v = { (_Float16)ra[i].x, (_Float16)ra[i].y,
                      (_Float16)ra[i].z, (_Float16)ra[i].w };
            *(v4h*)(&As[r * LDSS + k4 * 4]) = v;
        }
        // B: 4x4 in-register transpose -> [col][k], 4x ds_store_b64
        float t[4][4] = {
            { rb[0].x, rb[0].y, rb[0].z, rb[0].w },
            { rb[1].x, rb[1].y, rb[1].z, rb[1].w },
            { rb[2].x, rb[2].y, rb[2].z, rb[2].w },
            { rb[3].x, rb[3].y, rb[3].z, rb[3].w } };
#pragma unroll
        for (int c = 0; c < 4; ++c) {
            v4h v = { (_Float16)t[0][c], (_Float16)t[1][c],
                      (_Float16)t[2][c], (_Float16)t[3][c] };
            *(v4h*)(&Bs[(bcb + c) * LDSS + bkb]) = v;
        }
    };

    // Loop-invariant LDS fragment pointers
    const _Float16* ap  = &As[(wrow * 16 + l16) * LDSS + half * 8];
    const _Float16* bp0 = &Bs[(wcol * 64 + l16) * LDSS + half * 8];

    v8f acc[4] = {};   // 4 col-tiles of 16x16 f32 per wave

    loadA(aBase);
    loadB(bBase);

    for (int kc = 0; kc < NK; ++kc) {
        stage();
        __syncthreads();

        aBase += BK * 4;
        bBase += BK * BN * 4;
        if (kc + 1 < NK) { loadA(aBase); loadB(bBase); }  // prefetch next chunk

        // A fragment: lane holds row (wrow*16+l16), K in {half*8+0..7, 16+half*8+0..7}
        v8h alo = *(const v8h*)(ap);
        v8h ahi = *(const v8h*)(ap + 16);
        v16h af = __builtin_shufflevector(alo, ahi,
                    0,1,2,3,4,5,6,7,8,9,10,11,12,13,14,15);

#pragma unroll
        for (int tcol = 0; tcol < 4; ++tcol) {
            // B fragment: lane holds col (wcol*64+tcol*16+l16), K in {half*8+0..7, 16+half*8+0..7}
            const _Float16* bp = bp0 + tcol * 16 * LDSS;
            v8h blo = *(const v8h*)(bp);
            v8h bhi = *(const v8h*)(bp + 16);
            v16h bf = __builtin_shufflevector(blo, bhi,
                        0,1,2,3,4,5,6,7,8,9,10,11,12,13,14,15);

            acc[tcol] = __builtin_amdgcn_wmma_f32_16x16x32_f16(
                false, af, false, bf, (short)0, acc[tcol], false, false);
        }
        __syncthreads();
    }

    // Epilogue: C/D layout -> M = half*8 + r (VGPR r), N = l16 within tile
    const int mbase = row0 + wrow * 16 + half * 8;
    float sc[8];
    if (row_scale) {
#pragma unroll
        for (int r = 0; r < 8; ++r) sc[r] = row_scale[mbase + r];
    } else {
#pragma unroll
        for (int r = 0; r < 8; ++r) sc[r] = 1.0f;
    }
#pragma unroll
    for (int r = 0; r < 8; ++r) {
#pragma unroll
        for (int tcol = 0; tcol < 4; ++tcol) {
            const int n = wcol * 64 + tcol * 16 + l16;
            C[(size_t)(mbase + r) * BN + n] = acc[tcol][r] * sc[r];
        }
    }
}

extern "C" void kernel_launch(void* const* d_in, const int* in_sizes, int n_in,
                              void* d_out, int out_size, void* d_ws, size_t ws_size,
                              hipStream_t stream)
{
    (void)in_sizes; (void)n_in; (void)out_size; (void)ws_size;

    const int N = 8192, C_IN = 256, C_OUT = 128;

    const float* features     = (const float*)d_in[0];  // [N, C_IN]
    const float* wavelets     = (const float*)d_in[1];  // [N, N]
    const float* wavelets_inv = (const float*)d_in[2];  // [N, N]
    const float* weight       = (const float*)d_in[3];  // [C_IN, C_OUT]
    const float* filt         = (const float*)d_in[4];  // [N]
    float* out = (float*)d_out;                         // [N, C_OUT]

    float* transformed = (float*)d_ws;                       // [N, C_OUT] = 4 MB
    float* scaled      = transformed + (size_t)N * C_OUT;    // [N, C_OUT] = 4 MB

    dim3 blk(256);
    dim3 grd(N / BM);

    // 1) transformed = features @ W
    gwnn_gemm_f16<<<grd, blk, 0, stream>>>(features, C_IN, C_IN, weight,
                                           transformed, nullptr);
    // 2) scaled = diag(filt) * (wavelets_inv @ transformed)   (row scale fused)
    gwnn_gemm_f16<<<grd, blk, 0, stream>>>(wavelets_inv, N, N, transformed,
                                           scaled, filt);
    // 3) out = wavelets @ scaled
    gwnn_gemm_f16<<<grd, blk, 0, stream>>>(wavelets, N, N, scaled,
                                           out, nullptr);
}
